// BasicBlock_83227876262148
// MI455X (gfx1250) — 8-bit matrix operand order repaired and checked by reading, compiled, not run on hardware
//
#include <hip/hip_runtime.h>
#include <stdint.h>

typedef __attribute__((ext_vector_type(8))) int v8i;
typedef __attribute__((ext_vector_type(2))) unsigned int u2;
typedef __attribute__((ext_vector_type(4))) unsigned int u4;

typedef int av4i __attribute__((vector_size(16)));
typedef __attribute__((address_space(1))) av4i* gptr_v4i;
typedef __attribute__((address_space(3))) av4i* lptr_v4i;

#define BATCH 32
#define CCH   128
#define HWD   56
#define HWHW  3136
#define P_TOT 100352
#define X_ELEMS 12845056
#define W_ELEMS 147456
#define EPSBN 1e-5f

#define LWQ_STRIDE 292
#define LWQ_DWORDS (64 * LWQ_STRIDE)

#define OFF_SCAL 0
#define OFF_MEAN 1024
#define OFF_VAR  1536
#define OFF_BNS  2048
#define OFF_BNB  2560
#define OFF_PSUM 4096
#define OFF_PSQ  69632
#define OFF_WQ1  135168
#define OFF_WQ2  282624
#define OFF_XQ   430080
#define OFF_AQ   13275136
#define OFF_PRE  26120192

__global__ void init_ws_kernel(unsigned int* scal) {
    if (threadIdx.x < 4) scal[threadIdx.x] = 0u;
}

__global__ void absmax_kernel(const float* __restrict__ src, int n, unsigned int* dst) {
    __shared__ unsigned int red[256];
    unsigned int m = 0;
    for (int i = blockIdx.x * blockDim.x + threadIdx.x; i < n; i += gridDim.x * blockDim.x) {
        unsigned int b = __float_as_uint(fabsf(src[i]));
        m = m > b ? m : b;
    }
    red[threadIdx.x] = m;
    __syncthreads();
    for (int s = 128; s > 0; s >>= 1) {
        if ((int)threadIdx.x < s) red[threadIdx.x] = max(red[threadIdx.x], red[threadIdx.x + s]);
        __syncthreads();
    }
    if (threadIdx.x == 0) atomicMax(dst, red[0]);
}

__global__ void __launch_bounds__(256)
quant_x_kernel(const float* __restrict__ x, const unsigned int* scal,
               unsigned int* __restrict__ xq) {
    __shared__ signed char tile[64 * 132];
    int p0  = blockIdx.x * 64;
    int n   = p0 / HWHW;
    int hw0 = p0 - n * HWHW;
    float s   = __uint_as_float(*scal) + 1e-12f;
    float inv = 127.0f / s;
    int pj = threadIdx.x & 63;
    int c0 = threadIdx.x >> 6;
    for (int c = c0; c < CCH; c += 4) {
        float v = x[(n * CCH + c) * HWHW + hw0 + pj];
        int q = __float2int_rn(v * inv);
        q = min(127, max(-127, q));
        tile[pj * 132 + c] = (signed char)q;
    }
    __syncthreads();
    for (int i = threadIdx.x; i < 64 * 32; i += 256) {
        int pj2 = i >> 5;
        int cg  = i & 31;
        unsigned int pack = 0;
#pragma unroll
        for (int j = 0; j < 4; ++j)
            pack |= ((unsigned int)(unsigned char)tile[pj2 * 132 + cg * 4 + j]) << (8 * j);
        xq[(p0 + pj2) * 32 + cg] = pack;
    }
}

__global__ void quant_w_kernel(const float* __restrict__ w, const unsigned int* scal,
                               unsigned int* __restrict__ wq) {
    int idx = blockIdx.x * blockDim.x + threadIdx.x;
    if (idx >= CCH * 9 * (CCH / 4)) return;
    int o  = idx / (9 * 32);
    int t  = (idx / 32) % 9;
    int r  = t / 3, s2 = t % 3;
    int cg = idx & 31;
    float s   = __uint_as_float(*scal) + 1e-12f;
    float inv = 127.0f / s;
    unsigned int pack = 0;
#pragma unroll
    for (int j = 0; j < 4; ++j) {
        int c = cg * 4 + j;
        float v = w[((o * CCH + c) * 3 + r) * 3 + s2];
        int q = __float2int_rn(v * inv);
        q = min(127, max(-127, q));
        pack |= ((unsigned int)(q & 0xFF)) << (8 * j);
    }
    wq[idx] = pack;
}

__global__ void __launch_bounds__(256)
conv_wmma_kernel(const unsigned int* __restrict__ aq,
                 const unsigned int* __restrict__ wq,
                 const unsigned int* scalA, const unsigned int* scalB,
                 float* __restrict__ out)
{
    extern __shared__ unsigned int lwq[];

    int cobase = (blockIdx.x & 1) * 64;
    int mblk   = blockIdx.x >> 1;

    const unsigned int* wsrc = wq + cobase * (9 * 32);
    for (int i = threadIdx.x; i < 64 * 72; i += 256) {
        int co_l = i / 72;
        int ch   = i - co_l * 72;
        int dst  = co_l * LWQ_STRIDE + ch * 4;
        int src  = co_l * 288 + ch * 4;
#if __has_builtin(__builtin_amdgcn_global_load_async_to_lds_b128)
        __builtin_amdgcn_global_load_async_to_lds_b128(
            (gptr_v4i)(wsrc + src), (lptr_v4i)(lwq + dst), 0, 0);
#else
        *(u4*)(lwq + dst) = *(const u4*)(wsrc + src);
#endif
    }
#if __has_builtin(__builtin_amdgcn_global_load_async_to_lds_b128)
#if __has_builtin(__builtin_amdgcn_s_wait_asynccnt)
    __builtin_amdgcn_s_wait_asynccnt(0);
#else
    asm volatile("s_wait_asynccnt 0x0" ::: "memory");
#endif
#endif
    __syncthreads();

    int wave   = threadIdx.x >> 5;
    int lane   = threadIdx.x & 31;
    int half   = lane >> 4;
    int lane16 = lane & 15;
    int tile_m = mblk * 8 + wave;

    int p  = tile_m * 16 + lane16;
    int n  = p / HWHW;
    int rm = p - n * HWHW;
    int ho = rm / HWD;
    int wo = rm - ho * HWD;

    float sA = __uint_as_float(*scalA) + 1e-12f;
    float sB = __uint_as_float(*scalB) + 1e-12f;
    float scale = sA * sB * (1.0f / (127.0f * 127.0f));

    v8i zero = {};
    v8i acc[4];
#pragma unroll
    for (int t = 0; t < 4; ++t) acc[t] = zero;

#pragma unroll
    for (int r = 0; r < 3; ++r) {
        int hi = ho + r - 1;
#pragma unroll
        for (int s = 0; s < 3; ++s) {
            int wi = wo + s - 1;
            int tap = r * 3 + s;
            bool valid = (hi >= 0) & (hi < HWD) & (wi >= 0) & (wi < HWD);
            int abase = ((n * HWD + hi) * HWD + wi) * 32;
            if (valid && (hi + 1 < HWD))
                __builtin_prefetch(aq + abase + HWD * 32, 0, 1);
#pragma unroll
            for (int kc = 0; kc < 2; ++kc) {
                int ab = abase + kc * 16 + half * 4;
                u2 z2 = {};
                u2 A0 = valid ? *(const u2*)(aq + ab +  0) : z2;
                u2 A1 = valid ? *(const u2*)(aq + ab +  2) : z2;
                u2 A2 = valid ? *(const u2*)(aq + ab +  8) : z2;
                u2 A3 = valid ? *(const u2*)(aq + ab + 10) : z2;
                v8i a = {(int)A0.x, (int)A0.y, (int)A1.x, (int)A1.y,
                         (int)A2.x, (int)A2.y, (int)A3.x, (int)A3.y};
#pragma unroll
                for (int t = 0; t < 4; ++t) {
                    int bb = (t * 16 + lane16) * LWQ_STRIDE + tap * 32 + kc * 16 + half * 4;
                    u4 B0 = *(const u4*)(lwq + bb);
                    u4 B1 = *(const u4*)(lwq + bb + 8);
                    v8i b = {(int)B0.x, (int)B0.y, (int)B0.z, (int)B0.w,
                             (int)B1.x, (int)B1.y, (int)B1.z, (int)B1.w};
                    acc[t] = __builtin_amdgcn_wmma_i32_16x16x64_iu8(
                         true, a,  true, b, acc[t],
                         false,  false);
                }
            }
        }
    }
    int po = tile_m * 16;
#pragma unroll
    for (int t = 0; t < 4; ++t)
#pragma unroll
        for (int j = 0; j < 8; ++j) {
            int m = j + half * 8;
            out[(po + m) * CCH + cobase + t * 16 + lane16] = (float)acc[t][j] * scale;
        }
}

__global__ void bn_stats1_kernel(const float* __restrict__ pre,
                                 float* __restrict__ psum, float* __restrict__ psq) {
    int blk = blockIdx.x;
    int c   = threadIdx.x & 127;
    int ph  = threadIdx.x >> 7;
    float sum = 0.f, sq = 0.f;
    int pbase = blk * 784;
    for (int j = ph; j < 784; j += 2) {
        float v = pre[(pbase + j) * CCH + c];
        sum += v; sq += v * v;
    }
    __shared__ float rs[256], rq[256];
    rs[threadIdx.x] = sum; rq[threadIdx.x] = sq;
    __syncthreads();
    if (ph == 0) {
        psum[blk * CCH + c] = rs[c] + rs[c + 128];
        psq [blk * CCH + c] = rq[c] + rq[c + 128];
    }
}

__global__ void bn_stats2_kernel(const float* __restrict__ psum, const float* __restrict__ psq,
                                 float* mean, float* var) {
    int c = threadIdx.x;
    float s = 0.f, q = 0.f;
    for (int b = 0; b < 128; ++b) { s += psum[b * CCH + c]; q += psq[b * CCH + c]; }
    float m = s * (1.0f / (float)P_TOT);
    mean[c] = m;
    var[c]  = q * (1.0f / (float)P_TOT) - m * m;
}

__global__ void bn_coef_kernel(const float* mean, const float* var,
                               const float* gamma, const float* beta,
                               float* bns, float* bnb) {
    int c = threadIdx.x;
    if (c < CCH) {
        float sc = rsqrtf(var[c] + EPSBN) * gamma[c];
        bns[c] = sc;
        bnb[c] = beta[c] - mean[c] * sc;
    }
}

__global__ void bn_relu_absmax_kernel(const float* __restrict__ pre,
                                      const float* __restrict__ bns, const float* __restrict__ bnb,
                                      unsigned int* amax) {
    __shared__ unsigned int red[256];
    unsigned int m = 0;
    int total = P_TOT * CCH;
    for (int i = blockIdx.x * blockDim.x + threadIdx.x; i < total; i += gridDim.x * blockDim.x) {
        int c = i & (CCH - 1);
        float y = fmaxf(pre[i] * bns[c] + bnb[c], 0.0f);
        unsigned int b = __float_as_uint(y);
        m = m > b ? m : b;
    }
    red[threadIdx.x] = m;
    __syncthreads();
    for (int s = 128; s > 0; s >>= 1) {
        if ((int)threadIdx.x < s) red[threadIdx.x] = max(red[threadIdx.x], red[threadIdx.x + s]);
        __syncthreads();
    }
    if (threadIdx.x == 0) atomicMax(amax, red[0]);
}

__global__ void bn_relu_quant_kernel(const float* __restrict__ pre,
                                     const float* __restrict__ bns, const float* __restrict__ bnb,
                                     const unsigned int* amax, unsigned int* __restrict__ aq) {
    int idx = blockIdx.x * blockDim.x + threadIdx.x;
    if (idx >= P_TOT * (CCH / 4)) return;
    int cg = idx & 31;
    int c  = cg * 4;
    float s   = __uint_as_float(*amax) + 1e-12f;
    float inv = 127.0f / s;
    float4 v = ((const float4*)pre)[idx];
    float y0 = fmaxf(v.x * bns[c + 0] + bnb[c + 0], 0.0f);
    float y1 = fmaxf(v.y * bns[c + 1] + bnb[c + 1], 0.0f);
    float y2 = fmaxf(v.z * bns[c + 2] + bnb[c + 2], 0.0f);
    float y3 = fmaxf(v.w * bns[c + 3] + bnb[c + 3], 0.0f);
    int q0 = min(127, max(-127, __float2int_rn(y0 * inv)));
    int q1 = min(127, max(-127, __float2int_rn(y1 * inv)));
    int q2 = min(127, max(-127, __float2int_rn(y2 * inv)));
    int q3 = min(127, max(-127, __float2int_rn(y3 * inv)));
    aq[idx] = (unsigned int)(q0 & 0xFF) | ((unsigned int)(q1 & 0xFF) << 8) |
              ((unsigned int)(q2 & 0xFF) << 16) | ((unsigned int)(q3 & 0xFF) << 24);
}

__global__ void __launch_bounds__(256)
bn_add_relu_out_kernel(const float* __restrict__ pre, const float* __restrict__ x,
                       const float* __restrict__ bns, const float* __restrict__ bnb,
                       float* __restrict__ out) {
    __shared__ float tile[64 * 129];
    int p0  = blockIdx.x * 64;
    int n   = p0 / HWHW;
    int hw0 = p0 - n * HWHW;
    for (int i = threadIdx.x; i < 64 * 32; i += 256) {
        int pj = i >> 5;
        int cg = i & 31;
        float4 v = ((const float4*)pre)[(p0 + pj) * 32 + cg];
        tile[pj * 129 + cg * 4 + 0] = v.x;
        tile[pj * 129 + cg * 4 + 1] = v.y;
        tile[pj * 129 + cg * 4 + 2] = v.z;
        tile[pj * 129 + cg * 4 + 3] = v.w;
    }
    __syncthreads();
    int pj = threadIdx.x & 63;
    int c0 = threadIdx.x >> 6;
    for (int c = c0; c < CCH; c += 4) {
        int gi = (n * CCH + c) * HWHW + hw0 + pj;
        float y = tile[pj * 129 + c] * bns[c] + bnb[c] + x[gi];
        out[gi] = fmaxf(y, 0.0f);
    }
}

extern "C" void kernel_launch(void* const* d_in, const int* in_sizes, int n_in,
                              void* d_out, int out_size, void* d_ws, size_t ws_size,
                              hipStream_t stream) {
    const float* x      = (const float*)d_in[0];
    const float* w1     = (const float*)d_in[1];
    const float* gamma1 = (const float*)d_in[2];
    const float* beta1  = (const float*)d_in[3];
    const float* w2     = (const float*)d_in[4];
    const float* gamma2 = (const float*)d_in[5];
    const float* beta2  = (const float*)d_in[6];
    float* out = (float*)d_out;

    char* ws = (char*)d_ws;
    unsigned int* scal = (unsigned int*)(ws + OFF_SCAL);
    float* mean = (float*)(ws + OFF_MEAN);
    float* var  = (float*)(ws + OFF_VAR);
    float* bns  = (float*)(ws + OFF_BNS);
    float* bnb  = (float*)(ws + OFF_BNB);
    float* psum = (float*)(ws + OFF_PSUM);
    float* psq  = (float*)(ws + OFF_PSQ);
    unsigned int* wq1 = (unsigned int*)(ws + OFF_WQ1);
    unsigned int* wq2 = (unsigned int*)(ws + OFF_WQ2);
    unsigned int* xq  = (unsigned int*)(ws + OFF_XQ);
    unsigned int* aq  = (unsigned int*)(ws + OFF_AQ);
    float* pre = (float*)(ws + OFF_PRE);

    const size_t conv_lds = LWQ_DWORDS * sizeof(unsigned int);

    init_ws_kernel<<<1, 64, 0, stream>>>(scal);

    absmax_kernel<<<1024, 256, 0, stream>>>(x,  X_ELEMS, scal + 0);
    absmax_kernel<<<288,  256, 0, stream>>>(w1, W_ELEMS, scal + 1);
    absmax_kernel<<<288,  256, 0, stream>>>(w2, W_ELEMS, scal + 2);

    quant_x_kernel<<<P_TOT / 64, 256, 0, stream>>>(x, scal + 0, xq);
    quant_w_kernel<<<(CCH * 9 * 32 + 255) / 256, 256, 0, stream>>>(w1, scal + 1, wq1);
    quant_w_kernel<<<(CCH * 9 * 32 + 255) / 256, 256, 0, stream>>>(w2, scal + 2, wq2);

    conv_wmma_kernel<<<P_TOT / 16 / 8 * 2, 256, conv_lds, stream>>>(xq, wq1, scal + 0, scal + 1, pre);

    bn_stats1_kernel<<<128, 256, 0, stream>>>(pre, psum, psq);
    bn_stats2_kernel<<<1, 128, 0, stream>>>(psum, psq, mean, var);
    bn_coef_kernel<<<1, 128, 0, stream>>>(mean, var, gamma1, beta1, bns, bnb);
    bn_relu_absmax_kernel<<<2048, 256, 0, stream>>>(pre, bns, bnb, scal + 3);
    bn_relu_quant_kernel<<<(P_TOT * 32 + 255) / 256, 256, 0, stream>>>(pre, bns, bnb, scal + 3, aq);

    conv_wmma_kernel<<<P_TOT / 16 / 8 * 2, 256, conv_lds, stream>>>(aq, wq2, scal + 3, scal + 2, pre);

    bn_stats1_kernel<<<128, 256, 0, stream>>>(pre, psum, psq);
    bn_stats2_kernel<<<1, 128, 0, stream>>>(psum, psq, mean, var);
    bn_coef_kernel<<<1, 128, 0, stream>>>(mean, var, gamma2, beta2, bns, bnb);
    bn_add_relu_out_kernel<<<P_TOT / 64, 256, 0, stream>>>(pre, x, bns, bnb, out);
}
